// MultiLayerRelationalGATv2_35296041238629
// MI455X (gfx1250) — matrix operand order repaired and checked by reading, not compiled, not run on hardware
//
#include <hip/hip_runtime.h>
#include <hip/hip_bf16.h>

// ---------------------------------------------------------------------------
// Multi-layer relational GATv2 for MI455X (gfx1250, wave32).
//
// Dense transforms (x@Wl, x@Wr, residual) run as bf16 WMMA GEMMs
// (v_wmma_f32_16x16x32_bf16, fp32 accumulate).  The block's 64-column W slab
// (<=32KB) is staged in LDS ONCE, so the k-loop is a pure
// global_load(A) + ds_load(B) + wmma stream with no barriers.
// Edge attention is bandwidth/atomic bound and runs channel-coalesced with
// native global_atomic_add_f32.  LayerNorm uses an LDS tree reduction.
// ---------------------------------------------------------------------------

typedef __attribute__((ext_vector_type(16))) __bf16 v16bf;
typedef __attribute__((ext_vector_type(8)))  float  v8f;

union FragAB {
    uint4 u[2];
    v16bf v;
};

#define CH 256          // hidden/output channels (heads*c = 8*32)
#define HEADS 8
#define HC 32           // channels per head
#define NEG_SLOPE 0.2f
#define LN_EPS 1e-5f

// ------------------------------ utility kernels ----------------------------

__global__ void k_cast_bf16(const float* __restrict__ in, __bf16* __restrict__ out, int n) {
    int i = blockIdx.x * 256 + threadIdx.x;
    if (i < n) out[i] = (__bf16)in[i];
}

// W[k][n] (K x dout, row major fp32)  ->  Wt[n][k] (dout x K, row major bf16)
__global__ void k_wconv(const float* __restrict__ W, __bf16* __restrict__ Wt,
                        int K, int dout, int total) {
    int i = blockIdx.x * 256 + threadIdx.x;
    if (i < total) {
        int k = i / dout;
        int n = i - k * dout;
        Wt[(size_t)n * K + k] = (__bf16)W[i];
    }
}

__global__ void k_init_h(float* __restrict__ hb, const float* __restrict__ b0,
                         const float* __restrict__ b1, const float* __restrict__ b2, int total) {
    int i = blockIdx.x * 256 + threadIdx.x;
    if (i < total) {
        int c = i & (CH - 1);
        hb[i] = b0[c] + b1[c] + b2[c];
    }
}

__global__ void k_init_mden(float* __restrict__ m, float* __restrict__ den, int total) {
    int i = blockIdx.x * 256 + threadIdx.x;
    if (i < total) { m[i] = -1e30f; den[i] = 0.0f; }
}

// ------------------------------- WMMA GEMM ---------------------------------
// C[M x dout] = A[M x K](bf16) * W[K x dout] + bias;  Wt is W transposed
// (dout x K, bf16).  Block = 256 threads = 8 waves; each wave computes a
// 16x64 strip (4x v_wmma_f32_16x16x32_bf16 accumulators); block tile is
// 128 x 64.  The 64 x K W slab (max 32KB bf16) is staged in LDS once.
__global__ void __launch_bounds__(256)
k_gemm_bf16(const __bf16* __restrict__ A, const __bf16* __restrict__ Wt,
            const float* __restrict__ bias, float* __restrict__ C,
            int M, int K, int dout) {
    __shared__ __align__(16) __bf16 wslab[64 * 256];   // 32KB, K <= 256

    const int wid  = threadIdx.x >> 5;   // wave id 0..7 -> M subtile
    const int lane = threadIdx.x & 31;
    const int half = lane >> 4;          // 0: lanes 0-15, 1: lanes 16-31
    const int l16  = lane & 15;

    const int m0 = blockIdx.x * 128 + wid * 16;
    const int n0 = blockIdx.y * 64;

    // ---- stage the whole 64-column W slab for all K, once ----
    {
        const int chunksPerRow = K >> 3;            // 16B chunks per row
        const int nChunks = 64 * chunksPerRow;
        for (int c = threadIdx.x; c < nChunks; c += 256) {
            int r   = c / chunksPerRow;
            int sgm = c - r * chunksPerRow;
            *(uint4*)(&wslab[r * K + sgm * 8]) =
                *(const uint4*)(Wt + (size_t)(n0 + r) * K + sgm * 8);
        }
    }
    __syncthreads();

    int rowA = m0 + l16;
    if (rowA >= M) rowA = M - 1;         // clamp: keep EXEC all-1s for WMMA
    const __bf16* arow = A + (size_t)rowA * K;

    v8f acc[4];
#pragma unroll
    for (int j = 0; j < 4; ++j) acc[j] = (v8f){0.f,0.f,0.f,0.f,0.f,0.f,0.f,0.f};

#pragma unroll 4
    for (int k = 0; k < K; k += 32) {
        // A fragment (16-bit A 16x32 layout): lanes<16: K {0..7,16..23},
        // lanes>=16: K {8..15,24..31} of this k-step.
        FragAB a;
        const int ka = k + half * 8;
        a.u[0] = *(const uint4*)(arow + ka);
        a.u[1] = *(const uint4*)(arow + ka + 16);

        // All four B fragments first, so dscnt waits can be staggered.
        FragAB b[4];
        const __bf16* bp = &wslab[l16 * K + k + half * 8];
#pragma unroll
        for (int j = 0; j < 4; ++j) {
            b[j].u[0] = *(const uint4*)(bp + j * 16 * K);
            b[j].u[1] = *(const uint4*)(bp + j * 16 * K + 16);
        }
#pragma unroll
        for (int j = 0; j < 4; ++j) {
            acc[j] = __builtin_amdgcn_wmma_f32_16x16x32_bf16(
                false, a.v, false, b[j].v, (short)0, acc[j], false, false);
        }
    }

    // C/D layout: VGPR v, lanes 0-15 -> row m0+v, lanes 16-31 -> row m0+8+v;
    // column = n0 + 16*j + (lane&15).
#pragma unroll
    for (int j = 0; j < 4; ++j) {
        int col = n0 + 16 * j + l16;
        float bs = bias ? bias[col] : 0.0f;
#pragma unroll
        for (int v = 0; v < 8; ++v) {
            int row = m0 + v + half * 8;
            if (row < M) C[(size_t)row * dout + col] = acc[j][v] + bs;
        }
    }
}

// ------------------------------ edge kernels -------------------------------

// logits[e][h] = sum_c att[h][c] * leaky_relu(xl[src][h][c] + xr[dst][h][c])
__global__ void k_logits(const float* __restrict__ xl, const float* __restrict__ xr,
                         const int* __restrict__ src, const int* __restrict__ dst,
                         const float* __restrict__ att, float* __restrict__ logits,
                         int total) {
    int idx = blockIdx.x * 256 + threadIdx.x;
    if (idx >= total) return;
    int e = idx >> 3;
    int h = idx & 7;
    int s = src[e], d = dst[e];
    const float* pl = xl + (size_t)s * CH + h * HC;
    const float* pr = xr + (size_t)d * CH + h * HC;
    const float* pa = att + h * HC;
    float acc = 0.0f;
#pragma unroll
    for (int c = 0; c < HC; ++c) {
        float v = pl[c] + pr[c];
        v = (v > 0.0f) ? v : NEG_SLOPE * v;
        acc += v * pa[c];
    }
    logits[idx] = acc;
}

__device__ inline void atomicMaxF(float* addr, float val) {
    unsigned int* ua = (unsigned int*)addr;
    unsigned int old = *ua;
    while (__uint_as_float(old) < val) {
        unsigned int assumed = old;
        old = atomicCAS(ua, assumed, __float_as_uint(val));
        if (old == assumed) break;
    }
}

__global__ void k_segmax(const float* __restrict__ logits, const int* __restrict__ dst,
                         float* __restrict__ m, int total) {
    int idx = blockIdx.x * 256 + threadIdx.x;
    if (idx >= total) return;
    int e = idx >> 3, h = idx & 7;
    atomicMaxF(&m[dst[e] * HEADS + h], logits[idx]);
}

__global__ void k_expsum(const float* __restrict__ logits, const int* __restrict__ dst,
                         const float* __restrict__ m, float* __restrict__ den,
                         float* __restrict__ exv, int total) {
    int idx = blockIdx.x * 256 + threadIdx.x;
    if (idx >= total) return;
    int e = idx >> 3, h = idx & 7;
    float ex = __expf(logits[idx] - m[dst[e] * HEADS + h]);
    exv[idx] = ex;
    unsafeAtomicAdd(&den[dst[e] * HEADS + h], ex);
}

// hb[dst][cc] += (ex/den[dst][h]) * xl[src][cc]   (channel-parallel: E*256
// threads; gathers and atomics are fully coalesced within a wave)
__global__ void k_scatter(const float* __restrict__ exv, const float* __restrict__ den,
                          const float* __restrict__ xl, const int* __restrict__ src,
                          const int* __restrict__ dst, float* __restrict__ hb,
                          int total) {
    int idx = blockIdx.x * 256 + threadIdx.x;
    if (idx >= total) return;
    int e  = idx >> 8;          // edge
    int cc = idx & (CH - 1);    // channel 0..255
    int h  = cc >> 5;           // head
    int s = src[e], d = dst[e];
    float alpha = exv[e * HEADS + h] / den[d * HEADS + h];
    unsafeAtomicAdd(&hb[(size_t)d * CH + cc], alpha * xl[(size_t)s * CH + cc]);
}

// ------------------------- layernorm + relu (per node) ---------------------

__global__ void __launch_bounds__(256)
k_ln_relu(const float* __restrict__ hb, const float* __restrict__ res,
          const float* __restrict__ g, const float* __restrict__ b,
          float* __restrict__ out, int n) {
    __shared__ float red[256];
    int node = blockIdx.x;
    int c = threadIdx.x;
    float v = hb[(size_t)node * CH + c] + res[(size_t)node * CH + c];

    red[c] = v;
    __syncthreads();
    for (int s = 128; s > 0; s >>= 1) {
        if (c < s) red[c] += red[c + s];
        __syncthreads();
    }
    float mu = red[0] * (1.0f / CH);
    __syncthreads();

    float dv = v - mu;
    red[c] = dv * dv;
    __syncthreads();
    for (int s = 128; s > 0; s >>= 1) {
        if (c < s) red[c] += red[c + s];
        __syncthreads();
    }
    float var = red[0] * (1.0f / CH);

    float y = dv * rsqrtf(var + LN_EPS) * g[c] + b[c];
    out[(size_t)node * CH + c] = (y > 0.0f) ? y : 0.0f;
}

// ------------------------------- host driver -------------------------------

extern "C" void kernel_launch(void* const* d_in, const int* in_sizes, int n_in,
                              void* d_out, int out_size, void* d_ws, size_t ws_size,
                              hipStream_t stream) {
    const float* x0  = (const float*)d_in[0];
    const int*   eis[3] = { (const int*)d_in[1], (const int*)d_in[2], (const int*)d_in[3] };
    const int N = in_sizes[0] / 128;
    const int E = in_sizes[1] / 2;

    // Params: jax pytree flatten order (dict keys sorted):
    // per layer: convs[0..2]{Wl,Wr,att,bias,bl,br}, ln_b, ln_g, (res_W, res_b on layer0)
    const float *Wl[3][3], *Wr[3][3], *attp[3][3], *bias[3][3], *bl[3][3], *br[3][3];
    const float *lnb[3], *lng[3], *resW = nullptr, *resb = nullptr;
    int p = 4;
    for (int l = 0; l < 3; ++l) {
        for (int r = 0; r < 3; ++r) {
            Wl[l][r]   = (const float*)d_in[p++];
            Wr[l][r]   = (const float*)d_in[p++];
            attp[l][r] = (const float*)d_in[p++];
            bias[l][r] = (const float*)d_in[p++];
            bl[l][r]   = (const float*)d_in[p++];
            br[l][r]   = (const float*)d_in[p++];
        }
        lnb[l] = (const float*)d_in[p++];
        lng[l] = (const float*)d_in[p++];
        if (l == 0) {
            resW = (const float*)d_in[p++];
            resb = (const float*)d_in[p++];
        }
    }

    // Workspace carve-up (~350 MB total for N=50000, E=200000)
    char* base = (char*)d_ws;
    size_t off = 0;
    auto carve = [&](size_t bytes) -> void* {
        void* r = base + off;
        off = (off + bytes + 255) & ~(size_t)255;
        return r;
    };
    __bf16* xb     = (__bf16*)carve((size_t)N * CH * sizeof(__bf16));   // bf16 activations
    __bf16* wt     = (__bf16*)carve((size_t)CH * CH * sizeof(__bf16));  // transposed bf16 weight
    float*  xl     = (float*)carve((size_t)N * CH * sizeof(float));
    float*  xr     = (float*)carve((size_t)N * CH * sizeof(float));
    float*  hb     = (float*)carve((size_t)N * CH * sizeof(float));
    float*  xres   = (float*)carve((size_t)N * CH * sizeof(float));
    float*  xA     = (float*)carve((size_t)N * CH * sizeof(float));
    float*  xB     = (float*)carve((size_t)N * CH * sizeof(float));
    float*  logits = (float*)carve((size_t)E * HEADS * sizeof(float));
    float*  exv    = (float*)carve((size_t)E * HEADS * sizeof(float));
    float*  mb     = (float*)carve((size_t)N * HEADS * sizeof(float));
    float*  den    = (float*)carve((size_t)N * HEADS * sizeof(float));

    const int EH = E * HEADS;
    auto blocks = [](int n) { return (n + 255) / 256; };

    for (int l = 0; l < 3; ++l) {
        const int K = (l == 0) ? 128 : 256;
        const float* xin = (l == 0) ? x0 : ((l == 1) ? xA : xB);
        float* xout = (l == 2) ? (float*)d_out : ((l == 0) ? xA : xB);

        // cast input activations to bf16
        k_cast_bf16<<<blocks(N * K), 256, 0, stream>>>(xin, xb, N * K);

        dim3 gg((N + 127) / 128, CH / 64);

        // residual path
        const float* resptr;
        if (l == 0) {
            k_wconv<<<blocks(K * CH), 256, 0, stream>>>(resW, wt, K, CH, K * CH);
            k_gemm_bf16<<<gg, 256, 0, stream>>>(xb, wt, resb, xres, N, K, CH);
            resptr = xres;
        } else {
            resptr = xin;
        }

        // h initialized with the sum of the three per-relation output biases
        k_init_h<<<blocks(N * CH), 256, 0, stream>>>(hb, bias[l][0], bias[l][1], bias[l][2], N * CH);

        for (int r = 0; r < 3; ++r) {
            // xl = x@Wl + bl ; xr = x@Wr + br   (bf16 WMMA GEMMs)
            k_wconv<<<blocks(K * CH), 256, 0, stream>>>(Wl[l][r], wt, K, CH, K * CH);
            k_gemm_bf16<<<gg, 256, 0, stream>>>(xb, wt, bl[l][r], xl, N, K, CH);
            k_wconv<<<blocks(K * CH), 256, 0, stream>>>(Wr[l][r], wt, K, CH, K * CH);
            k_gemm_bf16<<<gg, 256, 0, stream>>>(xb, wt, br[l][r], xr, N, K, CH);

            const int* src = eis[r];
            const int* dst = eis[r] + E;

            k_logits<<<blocks(EH), 256, 0, stream>>>(xl, xr, src, dst, attp[l][r], logits, EH);
            k_init_mden<<<blocks(N * HEADS), 256, 0, stream>>>(mb, den, N * HEADS);
            k_segmax<<<blocks(EH), 256, 0, stream>>>(logits, dst, mb, EH);
            k_expsum<<<blocks(EH), 256, 0, stream>>>(logits, dst, mb, den, exv, EH);
            k_scatter<<<blocks(E * CH), 256, 0, stream>>>(exv, den, xl, src, dst, hb, E * CH);
        }

        k_ln_relu<<<N, 256, 0, stream>>>(hb, resptr, lng[l], lnb[l], xout, N);
    }
}
